// Network_79602923864332
// MI455X (gfx1250) — matrix operand order repaired and checked by reading, not compiled, not run on hardware
//
#include <hip/hip_runtime.h>
#include <hip/hip_bf16.h>
#include <math.h>

// CDNA5 / gfx1250, wave32. WMMA f32 <- f16 x f16, 16x16x32.
typedef __attribute__((ext_vector_type(16))) _Float16 v16h;
typedef __attribute__((ext_vector_type(8)))  _Float16 v8h;
typedef __attribute__((ext_vector_type(2)))  _Float16 v2h;
typedef __attribute__((ext_vector_type(8)))  float    v8f;

// ---------------------------------------------------------------------------
// Implicit-GEMM 9x9 stride-2 pad-4 conv + ReLU via V_WMMA_F32_16X16X32_F16.
// GEMM K-order: k = (ky*9+kx)*CIN + cin (cin fastest) so A and B fragments
// are contiguous 16B runs in LDS (ds_load_b128).
// Inter-layer activations are kept in f16 HWC ([iy][ix][cin]) so staging for
// CIN=8 layers is a pure 16B global->LDS copy (b128 in, b128 out) and
// conv1/conv2 stores are packed b128 writes.
//   IN_HWC : input is f16 HWC (else f32 NCHW)
//   OUT_HWC: output is f16 HWC (else f32 NCHW, for the head)
// One workgroup (8 wave32s) per (image, band of 8 output rows).
// ---------------------------------------------------------------------------
template<int CIN, int COUT, int WIN, bool IN_HWC, bool OUT_HWC>
__global__ __launch_bounds__(256)
void conv9s2_relu_wmma(const float* __restrict__ xf,
                       const _Float16* __restrict__ xh,
                       const float* __restrict__ w,
                       float* __restrict__ yf,
                       _Float16* __restrict__ yh) {
  constexpr int OHBAND = 8;
  constexpr int HIN = WIN;
  constexpr int OW  = WIN / 2;
  constexpr int OH  = WIN / 2;
  constexpr int KT  = CIN * 81;            // real GEMM K
  constexpr int CH  = (KT + 31) / 32;      // K chunks of 32
  constexpr int KTP = CH * 32;             // padded K
  constexpr int LG  = (CIN == 8) ? 3 : 1;  // log2(CIN)
  constexpr int SP  = KTP / CIN;           // padded spatial positions (96/84)
  constexpr int R   = 2 * OHBAND + 7;      // staged input rows incl. halo (23)
  constexpr int SWC = WIN + 7;             // staged cols (exact halo extent)
  constexpr int MT  = (COUT + 15) / 16;    // M tiles
  constexpr int TPR = OW / 16;             // N tiles per output row
  constexpr int TP  = (OHBAND * TPR) / 8;  // N tiles per wave per M tile
  static_assert(CIN == 2 || CIN == 8, "");
  static_assert(IN_HWC == (CIN == 8), "HWC input only for CIN=8 layers");
  static_assert((OHBAND * TPR) % 8 == 0 && TP >= 1, "");
  static_assert(OW % 16 == 0, "");

  __shared__ __align__(16) _Float16 lds_a[R * SWC * CIN];   // activations
  __shared__ __align__(16) _Float16 lds_w[MT * 16 * KTP];   // weights (padded)

  const int img  = blockIdx.y;
  const int band = blockIdx.x;
  const int oy0  = band * OHBAND;
  const int iy0  = 2 * oy0 - 4;

  // ---- stage zero-padded activation band into LDS ----
  if constexpr (IN_HWC) {
    // f16 HWC input: one 16B copy per (row, col) unit
    for (int u = threadIdx.x; u < R * SWC; u += 256) {
      int r  = u / SWC;
      int c  = u - r * SWC;
      int iy = iy0 + r;
      int ix = c - 4;
      v8h v;
      #pragma unroll
      for (int i = 0; i < 8; ++i) v[i] = (_Float16)0.0f;
      if (ix >= 0 && ix < WIN && iy >= 0 && iy < HIN)
        v = *(const v8h*)&xh[((size_t)(img * HIN + iy) * WIN + ix) * 8];
      *(v8h*)&lds_a[u * 8] = v;
    }
  } else {
    // f32 NCHW input (CIN=2): transpose 2 planes -> packed f16 pair
    for (int u = threadIdx.x; u < R * SWC; u += 256) {
      int r  = u / SWC;
      int c  = u - r * SWC;
      int iy = iy0 + r;
      int ix = c - 4;
      float v0 = 0.0f, v1 = 0.0f;
      if (ix >= 0 && ix < WIN && iy >= 0 && iy < HIN) {
        size_t base = ((size_t)(img * 2) * HIN + iy) * WIN + ix;
        v0 = xf[base];
        v1 = xf[base + (size_t)HIN * WIN];
      }
      v2h pr; pr[0] = (_Float16)v0; pr[1] = (_Float16)v1;
      *(v2h*)&lds_a[u * 2] = pr;
    }
  }
  // ---- stage weights in fragment K-order (vectorized over cin) ----
  for (int u = threadIdx.x; u < MT * 16 * SP; u += 256) {
    int m = u / SP;
    int s = u - m * SP;              // s = ky*9 + kx (padded to SP)
    if constexpr (CIN == 8) {
      v8h v;
      #pragma unroll
      for (int cin = 0; cin < 8; ++cin) {
        float f = 0.0f;
        if (m < COUT && s < 81) f = w[(m * 8 + cin) * 81 + s];
        v[cin] = (_Float16)f;
      }
      *(v8h*)&lds_w[m * KTP + s * 8] = v;
    } else {
      v2h v;
      #pragma unroll
      for (int cin = 0; cin < 2; ++cin) {
        float f = 0.0f;
        if (m < COUT && s < 81) f = w[(m * 2 + cin) * 81 + s];
        v[cin] = (_Float16)f;
      }
      *(v2h*)&lds_w[m * KTP + s * 2] = v;
    }
  }
  __syncthreads();

  const int lane = threadIdx.x & 31;
  const int wave = threadIdx.x >> 5;
  const int half = lane >> 4;        // lane halves carry K halves (contiguous k order used here)
  const int idx  = lane & 15;        // A: row M, B: col N

  // per-tile activation base offsets (element index into lds_a)
  int abase_t[TP];
  #pragma unroll
  for (int t = 0; t < TP; ++t) {
    const int s   = wave + t * 8;
    const int oyl = s / TPR;
    const int ox  = (s % TPR) * 16 + idx;
    abase_t[t] = (2 * oyl * SWC + 2 * ox) * CIN;
  }

  for (int mt = 0; mt < MT; ++mt) {
    v8f acc[TP];
    #pragma unroll
    for (int t = 0; t < TP; ++t) {
      #pragma unroll
      for (int i = 0; i < 8; ++i) acc[t][i] = 0.0f;
    }

    const _Float16* wrow = &lds_w[(mt * 16 + idx) * KTP];

    for (int ch = 0; ch < CH; ++ch) {
      const int kbase = ch * 32;

      // ---- A fragment: elements e <-> k = kbase + 16*half + e ----
      const _Float16* wp = wrow + kbase + 16 * half;
      v8h alo = *(const v8h*)(wp);
      v8h ahi = *(const v8h*)(wp + 8);
      v16h a = __builtin_shufflevector(alo, ahi,
                 0, 1, 2, 3, 4, 5, 6, 7, 8, 9, 10, 11, 12, 13, 14, 15);

      // ---- B fragment: elements e <-> k = kbase + 16*half + e ----
      const int sbase = (kbase + 16 * half) >> LG;   // spatial base index

      if constexpr (CIN == 8) {
        int s0 = sbase;     if (s0 > 80) s0 = 80;
        int s1 = sbase + 1; if (s1 > 80) s1 = 80;
        const int ky0 = s0 / 9, kx0 = s0 - 9 * ky0;
        const int ky1 = s1 / 9, kx1 = s1 - 9 * ky1;
        const int soff0 = (ky0 * SWC + kx0) * 8;
        const int soff1 = (ky1 * SWC + kx1) * 8;
        #pragma unroll
        for (int t = 0; t < TP; ++t) {
          v8h blo = *(const v8h*)&lds_a[abase_t[t] + soff0];
          v8h bhi = *(const v8h*)&lds_a[abase_t[t] + soff1];
          v16h b = __builtin_shufflevector(blo, bhi,
                     0, 1, 2, 3, 4, 5, 6, 7, 8, 9, 10, 11, 12, 13, 14, 15);
          acc[t] = __builtin_amdgcn_wmma_f32_16x16x32_f16(
                       false, a, false, b, (short)0, acc[t], false, false);
        }
      } else {  // CIN == 2
        int soff[8];
        #pragma unroll
        for (int g = 0; g < 8; ++g) {
          int s = sbase + g; if (s > 80) s = 80;
          int ky = s / 9, kx = s - 9 * ky;
          soff[g] = (ky * SWC + kx) * 2;
        }
        #pragma unroll
        for (int t = 0; t < TP; ++t) {
          v16h b;
          #pragma unroll
          for (int g = 0; g < 8; ++g) {
            v2h pr = *(const v2h*)&lds_a[abase_t[t] + soff[g]];
            b[2 * g + 0] = pr[0];
            b[2 * g + 1] = pr[1];
          }
          acc[t] = __builtin_amdgcn_wmma_f32_16x16x32_f16(
                       false, a, false, b, (short)0, acc[t], false, false);
        }
      }
    }

    // ---- store with ReLU; C/D layout: VGPR r -> M = r + 8*half ----
    #pragma unroll
    for (int t = 0; t < TP; ++t) {
      const int s   = wave + t * 8;
      const int oyl = s / TPR;
      const int oxb = (s % TPR) * 16;
      const int oy  = oy0 + oyl;
      if constexpr (OUT_HWC) {
        // lane idx = pixel; its 8 rows = couts 8*half..8*half+7 -> one b128
        if (8 * half < COUT) {
          v8h o;
          #pragma unroll
          for (int r = 0; r < 8; ++r)
            o[r] = (_Float16)fmaxf(acc[t][r], 0.0f);
          size_t pix = (size_t)(img * OH + oy) * OW + oxb + idx;
          *(v8h*)&yh[pix * COUT + 8 * half] = o;
        }
      } else {
        #pragma unroll
        for (int r = 0; r < 8; ++r) {
          int cout = mt * 16 + r + 8 * half;
          if (cout < COUT) {
            yf[((size_t)(img * COUT + cout) * OH + oy) * OW + oxb + idx] =
                fmaxf(acc[t][r], 0.0f);
          }
        }
      }
    }
  }
}

// ---------------------------------------------------------------------------
// Head: per-channel spatial max -> threshold at 0.75*max -> masked write,
// mean pool, 32->2 linear, log_softmax. One block per image, 8 wave32s.
// ---------------------------------------------------------------------------
__global__ __launch_bounds__(256)
void head_kernel(const float* __restrict__ conv3,
                 const float* __restrict__ wfc,
                 float* __restrict__ out_y,
                 float* __restrict__ out_mask,
                 float* __restrict__ out_pool) {
  const int img  = blockIdx.x;
  const int lane = threadIdx.x & 31;
  const int wave = threadIdx.x >> 5;

  __shared__ float pool_s[32];

  #pragma unroll
  for (int cc = 0; cc < 4; ++cc) {
    const int ch = wave * 4 + cc;
    const float* p = conv3 + ((size_t)img * 32 + ch) * 256;

    float v[8];
    float mx = -1e30f;
    #pragma unroll
    for (int j = 0; j < 8; ++j) {
      v[j] = p[lane + 32 * j];
      mx = fmaxf(mx, v[j]);
    }
    #pragma unroll
    for (int s = 1; s < 32; s <<= 1) mx = fmaxf(mx, __shfl_xor(mx, s, 32));

    const float th = mx * 0.75f;
    float sum = 0.0f;
    #pragma unroll
    for (int j = 0; j < 8; ++j) {
      float m = (v[j] < th) ? 0.0f : v[j];
      out_mask[((size_t)img * 32 + ch) * 256 + lane + 32 * j] = m;
      sum += m;
    }
    #pragma unroll
    for (int s = 1; s < 32; s <<= 1) sum += __shfl_xor(sum, s, 32);

    if (lane == 0) {
      float pl = sum * (1.0f / 256.0f);
      pool_s[ch] = pl;
      out_pool[(size_t)img * 32 + ch] = pl;
    }
  }
  __syncthreads();

  if (threadIdx.x == 0) {
    float l0 = 0.0f, l1 = 0.0f;
    #pragma unroll
    for (int ch = 0; ch < 32; ++ch) {
      l0 += pool_s[ch] * wfc[ch];
      l1 += pool_s[ch] * wfc[32 + ch];
    }
    float m   = fmaxf(l0, l1);
    float lse = m + logf(expf(l0 - m) + expf(l1 - m));
    out_y[(size_t)img * 2 + 0] = l0 - lse;
    out_y[(size_t)img * 2 + 1] = l1 - lse;
  }
}

// ---------------------------------------------------------------------------
// kernel_launch: x -> conv1(f16 HWC) -> conv2(f16 HWC) -> conv3(f32 NCHW)
//                -> head (outputs)
// d_out layout (return order, flat): y [1024*2] | conv3_masked [1024*32*256]
//                                    | pool [1024*32]
// ---------------------------------------------------------------------------
extern "C" void kernel_launch(void* const* d_in, const int* in_sizes, int n_in,
                              void* d_out, int out_size, void* d_ws, size_t ws_size,
                              hipStream_t stream) {
  (void)in_sizes; (void)n_in; (void)out_size; (void)ws_size;

  const float* x   = (const float*)d_in[0];  // [1024, 2, 128, 128]
  const float* w1  = (const float*)d_in[1];  // [8, 2, 9, 9]
  const float* w2  = (const float*)d_in[2];  // [8, 8, 9, 9]
  const float* w3  = (const float*)d_in[3];  // [32, 8, 9, 9]
  const float* wfc = (const float*)d_in[4];  // [2, 32]

  // Workspace: act1 f16 HWC (64MB) | act2 f16 HWC (16MB) | c3 f32 NCHW (32MB)
  char* wsb = (char*)d_ws;
  _Float16* act1 = (_Float16*)wsb;                         // 1024*64*64*8 f16
  _Float16* act2 = (_Float16*)(wsb + (size_t)67108864);    // 1024*32*32*8 f16
  float*    c3   = (float*)(wsb + (size_t)67108864 + (size_t)16777216);

  float* y    = (float*)d_out;                   // 2048
  float* mask = y + 2048;                        // 8388608
  float* pool = mask + (size_t)8388608;          // 32768

  // conv1: 2->8, 128x128 -> 64x64 (f32 NCHW in, f16 HWC out)
  conv9s2_relu_wmma<2, 8, 128, false, true>
      <<<dim3(8, 1024), 256, 0, stream>>>(x, nullptr, w1, nullptr, act1);
  // conv2: 8->8, 64x64 -> 32x32 (f16 HWC in, f16 HWC out)
  conv9s2_relu_wmma<8, 8, 64, true, true>
      <<<dim3(4, 1024), 256, 0, stream>>>(nullptr, act1, w2, nullptr, act2);
  // conv3: 8->32, 32x32 -> 16x16 (f16 HWC in, f32 NCHW out for the head)
  conv9s2_relu_wmma<8, 32, 32, true, false>
      <<<dim3(2, 1024), 256, 0, stream>>>(nullptr, act2, w3, c3, nullptr);
  // head: mask/pool/fc/log_softmax
  head_kernel<<<dim3(1024), 256, 0, stream>>>(c3, wfc, y, mask, pool);
}
